// WindowedMultiHeadAttention_31078383354082
// MI455X (gfx1250) — matrix operand order repaired and checked by reading, not compiled, not run on hardware
//
#include <hip/hip_runtime.h>
#include <hip/hip_bf16.h>

#define HEADS 16
#define DIM   512
#define NW    64
#define NTOK  49
#define HD    32
#define BWIN  2048
#define MTOT  (BWIN * NTOK)          // 100352 (divisible by 128)
#define QK_SCALE 0.17677669529663687f // 1/sqrt(32)

typedef unsigned short u16;
typedef unsigned int   u32;
typedef __attribute__((ext_vector_type(16))) __bf16 v16bf;
typedef __attribute__((ext_vector_type(2)))  __bf16 v2bf;
typedef __attribute__((ext_vector_type(8)))  float  v8f;

union FragB { v16bf v; u32 u[8]; };

// LDS strides (in u16 elements) chosen to spread 64 banks, keep u32/uint4 align
#define STRA 40   // for Nx32 tiles (80B rows -> 20-bank stride)
#define STRP 72   // for 64x64 P tile (144B rows -> 36-bank stride)

// ---- f32 -> bf16 (RNE) ----
#if __has_builtin(__builtin_amdgcn_cvt_pk_bf16_f32)
__device__ __forceinline__ u32 pack2(float a, float b) {
  union { v2bf v; u32 u; } c;
  c.v = __builtin_amdgcn_cvt_pk_bf16_f32(a, b);
  return c.u;
}
__device__ __forceinline__ u16 f2bf(float f) { return (u16)pack2(f, f); }
#else
__device__ __forceinline__ u16 f2bf(float f) {
  union { __bf16 b; u16 u; } c;
  c.b = (__bf16)f;              // let the backend pick a native v_cvt
  return c.u;
}
__device__ __forceinline__ u32 pack2(float a, float b) {
  return (u32)f2bf(a) | ((u32)f2bf(b) << 16);
}
#endif

// ---- CDNA5 async global -> LDS copy (16B per lane), ASYNCcnt tracked ----
__device__ __forceinline__ void async_b128(void* lds_ptr, const void* gptr) {
  // low 32 bits of a flat LDS-aperture address == LDS byte offset
  u32 l = (u32)(unsigned long long)lds_ptr;
  asm volatile("global_load_async_to_lds_b128 %0, %1, off"
               :: "v"(l), "v"(gptr) : "memory");
}
__device__ __forceinline__ void wait_async0() {
#if __has_builtin(__builtin_amdgcn_s_wait_asynccnt)
  __builtin_amdgcn_s_wait_asynccnt(0);
#else
  asm volatile("s_wait_asynccnt 0" ::: "memory");
#endif
}

// ---------------------------------------------------------------------------
// Kernel: elementwise f32 -> bf16 (vectorized x4)
// ---------------------------------------------------------------------------
__global__ void cvt_bf16_kernel(const float* __restrict__ in,
                                u16* __restrict__ outp, int n4) {
  int i = blockIdx.x * 256 + threadIdx.x;
  if (i >= n4) return;
  float4 f = ((const float4*)in)[i];
  uint2 o;
  o.x = pack2(f.x, f.y);
  o.y = pack2(f.z, f.w);
  ((uint2*)outp)[i] = o;
}

// ---------------------------------------------------------------------------
// Kernel: f32 [K,N] -> bf16 transposed [N,K] (32x32 LDS tiles)
// ---------------------------------------------------------------------------
__global__ __launch_bounds__(256) void cvtT_bf16_kernel(
    const float* __restrict__ in, u16* __restrict__ outp, int K, int N) {
  __shared__ float t[32][33];
  int kb = blockIdx.x * 32, nb = blockIdx.y * 32;
  int tx = threadIdx.x & 31, ty = threadIdx.x >> 5;   // 32 x 8
  #pragma unroll
  for (int r = ty; r < 32; r += 8)
    t[r][tx] = in[(size_t)(kb + r) * N + nb + tx];
  __syncthreads();
  #pragma unroll
  for (int r = ty; r < 32; r += 8)
    outp[(size_t)(nb + r) * K + kb + tx] = f2bf(t[tx][r]);
}

// ---------------------------------------------------------------------------
// Kernel: bias_h[h][m][n] = bias_table[rel_idx[m*49+n]][h]
// ---------------------------------------------------------------------------
__global__ void bias_mat_kernel(const float* __restrict__ bt,
                                const int* __restrict__ ridx,
                                float* __restrict__ bias_h) {
  int i = blockIdx.x * 256 + threadIdx.x;
  if (i >= HEADS * NTOK * NTOK) return;
  int h  = i / (NTOK * NTOK);
  int mn = i % (NTOK * NTOK);
  bias_h[i] = bt[ridx[mn] * HEADS + h];
}

// ---------------------------------------------------------------------------
// Kernel 1: QKV GEMM  xb[100352,512](bf16) @ wbT^T + b -> bf16 scatter
// 128x64 block tile, 4 waves, wave = 32x64 strip; async double-buffered LDS.
// wbT is [1536,512] (pre-transposed weights). q pre-scaled by 1/sqrt(hd).
// ---------------------------------------------------------------------------
__global__ __launch_bounds__(128) void qkv_gemm_kernel(
    const u16* __restrict__ xb, const u16* __restrict__ wbT,
    const float* __restrict__ bias, u16* __restrict__ qkv_ws) {
  __shared__ __align__(16) u16 a_s[2][128 * STRA];
  __shared__ __align__(16) u16 b_s[2][64 * STRA];   // [n][k]

  const int tileM = blockIdx.x % (MTOT / 128);
  const int tileN = blockIdx.x / (MTOT / 128);
  const int m0 = tileM * 128, n0 = tileN * 64;
  const int tid = threadIdx.x, wv = tid >> 5, lane = tid & 31;
  const int lm = lane & 15, lh = lane >> 4;

  const int sr = tid >> 2;          // staging row 0..31
  const int sc = (tid & 3) * 8;     // staging col 0,8,16,24 (u16 units)

  // async-copy K-step 0 into buffer 0
  #pragma unroll
  for (int it = 0; it < 4; ++it)
    async_b128(&a_s[0][(it * 32 + sr) * STRA + sc],
               xb + (size_t)(m0 + it * 32 + sr) * DIM + sc);
  #pragma unroll
  for (int it = 0; it < 2; ++it)
    async_b128(&b_s[0][(it * 32 + sr) * STRA + sc],
               wbT + (size_t)(n0 + it * 32 + sr) * DIM + sc);
  wait_async0();
  __syncthreads();

  v8f zero = {};
  v8f acc[2][4] = {{zero, zero, zero, zero}, {zero, zero, zero, zero}};

  for (int ks = 0; ks < DIM / 32; ++ks) {
    const int cur = ks & 1, nxt = cur ^ 1;
    const int kn = (ks + 1) * 32;
    if (kn < DIM) {   // kick off next K-step copies; they overlap the WMMAs
      #pragma unroll
      for (int it = 0; it < 4; ++it)
        async_b128(&a_s[nxt][(it * 32 + sr) * STRA + sc],
                   xb + (size_t)(m0 + it * 32 + sr) * DIM + kn + sc);
      #pragma unroll
      for (int it = 0; it < 2; ++it)
        async_b128(&b_s[nxt][(it * 32 + sr) * STRA + sc],
                   wbT + (size_t)(n0 + it * 32 + sr) * DIM + kn + sc);
    }
    FragB a[2];
    #pragma unroll
    for (int mt = 0; mt < 2; ++mt) {
      const u16* ap = &a_s[cur][(wv * 32 + mt * 16 + lm) * STRA];
      int kb = lh * 16;
      #pragma unroll
      for (int i = 0; i < 4; ++i) {
        a[mt].u[i]     = *(const u32*)(ap + kb + 2 * i);
        a[mt].u[4 + i] = *(const u32*)(ap + kb + 8 + 2 * i);
      }
    }
    #pragma unroll
    for (int nj = 0; nj < 4; ++nj) {
      FragB b;
      const u16* bp = &b_s[cur][(nj * 16 + lm) * STRA];
      int kb2 = lh * 16;
      #pragma unroll
      for (int i = 0; i < 8; ++i) b.u[i] = *(const u32*)(bp + kb2 + 2 * i);
      #pragma unroll
      for (int mt = 0; mt < 2; ++mt)
        acc[mt][nj] = __builtin_amdgcn_wmma_f32_16x16x32_bf16(
            false, a[mt].v, false, b.v, (short)0, acc[mt][nj], false, false);
    }
    wait_async0();
    __syncthreads();
  }

  // epilogue: bias, scale q, scatter to [b,h,part,n_tok,d] bf16
  #pragma unroll
  for (int mt = 0; mt < 2; ++mt) {
    #pragma unroll
    for (int i = 0; i < 8; ++i) {
      int gm = m0 + wv * 32 + mt * 16 + i + ((lane & 16) ? 8 : 0);
      int bidx = gm / NTOK;
      int ntok = gm - bidx * NTOK;
      #pragma unroll
      for (int nj = 0; nj < 4; ++nj) {
        int gc = n0 + nj * 16 + lm;
        float v = acc[mt][nj][i] + bias[gc];
        int part = gc >> 9;
        int rem  = gc & 511;
        int h = rem >> 5, d = rem & 31;
        if (part == 0) v *= QK_SCALE;
        qkv_ws[((size_t)((bidx * HEADS + h) * 3 + part) * NTOK + ntok) * HD + d] = f2bf(v);
      }
    }
  }
}

// ---------------------------------------------------------------------------
// Kernel 2: fused attention per (window, head). 49 padded to 64.
// ---------------------------------------------------------------------------
__global__ __launch_bounds__(128) void attn_kernel(
    const u16* __restrict__ qkv_ws, const float* __restrict__ bias_h,
    const float* __restrict__ mask, u16* __restrict__ ao) {
  __shared__ __align__(16) u16 q_s[64 * STRA];
  __shared__ __align__(16) u16 k_s[64 * STRA];
  __shared__ __align__(16) u16 v_s[64 * STRA];
  __shared__ __align__(16) u16 p_s[64 * STRP];

  const int bw   = blockIdx.x >> 4;
  const int h    = blockIdx.x & 15;
  const int wmsk = bw & (NW - 1);
  const size_t base = (size_t)((bw * HEADS + h) * 3) * (NTOK * HD);
  const u32* qg = (const u32*)(qkv_ws + base);
  const u32* kg = (const u32*)(qkv_ws + base + NTOK * HD);
  const u32* vg = (const u32*)(qkv_ws + base + 2 * NTOK * HD);

  const int tid = threadIdx.x;
  for (int i = tid; i < 1024; i += 128) {   // 1024 u32 = 64 rows * 16 u32
    int row = i >> 4, cu = i & 15;
    u32 qv = (row < NTOK) ? qg[row * 16 + cu] : 0u;
    u32 kv = (row < NTOK) ? kg[row * 16 + cu] : 0u;
    u32 vv = (row < NTOK) ? vg[row * 16 + cu] : 0u;
    ((u32*)(q_s + row * STRA))[cu] = qv;
    ((u32*)(k_s + row * STRA))[cu] = kv;
    ((u32*)(v_s + row * STRA))[cu] = vv;
  }
  __syncthreads();

  const int wv = tid >> 5, lane = tid & 31;
  const int lm = lane & 15, lh = lane >> 4;
  const int m0 = wv * 16;

  // S = q . k^T   (A = q 16x32, B[d][n] = k[n][d] 32x16)
  v8f s[4];
  {
    FragB a;
    const u16* ap = &q_s[(m0 + lm) * STRA];
    int kb = lh * 16;
    #pragma unroll
    for (int i = 0; i < 4; ++i) {
      a.u[i]     = *(const u32*)(ap + kb + 2 * i);
      a.u[4 + i] = *(const u32*)(ap + kb + 8 + 2 * i);
    }
    #pragma unroll
    for (int nj = 0; nj < 4; ++nj) {
      FragB b;
      const u16* bp = &k_s[(nj * 16 + lm) * STRA];
      int kb2 = lh * 16;
      #pragma unroll
      for (int i = 0; i < 8; ++i) b.u[i] = *(const u32*)(bp + kb2 + 2 * i);
      v8f z = {};
      s[nj] = __builtin_amdgcn_wmma_f32_16x16x32_bf16(
          false, a.v, false, b.v, (short)0, z, false, false);
    }
  }

  // bias + mask + softmax (row m lives in one 16-lane half, reg i)
  #pragma unroll
  for (int i = 0; i < 8; ++i) {
    int m = m0 + i + ((lane & 16) ? 8 : 0);
    float e[4];
    if (m < NTOK) {
      const float* bp = bias_h + (size_t)(h * NTOK + m) * NTOK;
      const float* mp = mask + ((size_t)wmsk * NTOK + m) * NTOK;
      #pragma unroll
      for (int nj = 0; nj < 4; ++nj) {
        int n = nj * 16 + lm;
        e[nj] = (n < NTOK) ? (s[nj][i] + bp[n] + mp[n]) : -1e30f;
      }
    } else {
      #pragma unroll
      for (int nj = 0; nj < 4; ++nj) e[nj] = 0.0f;   // harmless uniform row
    }
    float mx = fmaxf(fmaxf(e[0], e[1]), fmaxf(e[2], e[3]));
    #pragma unroll
    for (int off = 1; off < 16; off <<= 1) mx = fmaxf(mx, __shfl_xor(mx, off, 16));
    float sum = 0.0f;
    #pragma unroll
    for (int nj = 0; nj < 4; ++nj) { e[nj] = __expf(e[nj] - mx); sum += e[nj]; }
    #pragma unroll
    for (int off = 1; off < 16; off <<= 1) sum += __shfl_xor(sum, off, 16);
    float inv = 1.0f / sum;
    #pragma unroll
    for (int nj = 0; nj < 4; ++nj)
      p_s[m * STRP + nj * 16 + lm] = f2bf(e[nj] * inv);
  }
  __syncthreads();

  // O = P @ V  (K = 64 in two 32-steps; N = 32 in two 16-tiles)
  v8f zero = {};
  v8f o[2] = {zero, zero};
  #pragma unroll
  for (int ks = 0; ks < 2; ++ks) {
    FragB a;
    const u16* ap = &p_s[(m0 + lm) * STRP + ks * 32];
    int kb = lh * 16;
    #pragma unroll
    for (int i = 0; i < 4; ++i) {
      a.u[i]     = *(const u32*)(ap + kb + 2 * i);
      a.u[4 + i] = *(const u32*)(ap + kb + 8 + 2 * i);
    }
    #pragma unroll
    for (int nt = 0; nt < 2; ++nt) {
      FragB b;
      int kb2 = ks * 32 + lh * 16;
      int n = nt * 16 + lm;
      #pragma unroll
      for (int i = 0; i < 8; ++i)
        b.u[i] = (u32)v_s[(kb2 + 2 * i) * STRA + n] |
                 ((u32)v_s[(kb2 + 2 * i + 1) * STRA + n] << 16);
      o[nt] = __builtin_amdgcn_wmma_f32_16x16x32_bf16(
          false, a.v, false, b.v, (short)0, o[nt], false, false);
    }
  }

  #pragma unroll
  for (int nt = 0; nt < 2; ++nt) {
    #pragma unroll
    for (int i = 0; i < 8; ++i) {
      int m = m0 + i + ((lane & 16) ? 8 : 0);
      if (m < NTOK)
        ao[((size_t)(bw * NTOK + m)) * DIM + h * HD + nt * 16 + lm] = f2bf(o[nt][i]);
    }
  }
}

// ---------------------------------------------------------------------------
// Kernel 3: proj GEMM  ao[100352,512](bf16) @ pwbT^T + b -> f32 out
// 128x64 block tile, async double-buffered LDS. pwbT is [512,512] transposed.
// ---------------------------------------------------------------------------
__global__ __launch_bounds__(128) void proj_gemm_kernel(
    const u16* __restrict__ ain, const u16* __restrict__ pwbT,
    const float* __restrict__ bias, float* __restrict__ out) {
  __shared__ __align__(16) u16 a_s[2][128 * STRA];
  __shared__ __align__(16) u16 b_s[2][64 * STRA];

  const int tileM = blockIdx.x % (MTOT / 128);
  const int tileN = blockIdx.x / (MTOT / 128);
  const int m0 = tileM * 128, n0 = tileN * 64;
  const int tid = threadIdx.x, wv = tid >> 5, lane = tid & 31;
  const int lm = lane & 15, lh = lane >> 4;

  const int sr = tid >> 2;
  const int sc = (tid & 3) * 8;

  #pragma unroll
  for (int it = 0; it < 4; ++it)
    async_b128(&a_s[0][(it * 32 + sr) * STRA + sc],
               ain + (size_t)(m0 + it * 32 + sr) * DIM + sc);
  #pragma unroll
  for (int it = 0; it < 2; ++it)
    async_b128(&b_s[0][(it * 32 + sr) * STRA + sc],
               pwbT + (size_t)(n0 + it * 32 + sr) * DIM + sc);
  wait_async0();
  __syncthreads();

  v8f zero = {};
  v8f acc[2][4] = {{zero, zero, zero, zero}, {zero, zero, zero, zero}};

  for (int ks = 0; ks < DIM / 32; ++ks) {
    const int cur = ks & 1, nxt = cur ^ 1;
    const int kn = (ks + 1) * 32;
    if (kn < DIM) {
      #pragma unroll
      for (int it = 0; it < 4; ++it)
        async_b128(&a_s[nxt][(it * 32 + sr) * STRA + sc],
                   ain + (size_t)(m0 + it * 32 + sr) * DIM + kn + sc);
      #pragma unroll
      for (int it = 0; it < 2; ++it)
        async_b128(&b_s[nxt][(it * 32 + sr) * STRA + sc],
                   pwbT + (size_t)(n0 + it * 32 + sr) * DIM + kn + sc);
    }
    FragB a[2];
    #pragma unroll
    for (int mt = 0; mt < 2; ++mt) {
      const u16* ap = &a_s[cur][(wv * 32 + mt * 16 + lm) * STRA];
      int kb = lh * 16;
      #pragma unroll
      for (int i = 0; i < 4; ++i) {
        a[mt].u[i]     = *(const u32*)(ap + kb + 2 * i);
        a[mt].u[4 + i] = *(const u32*)(ap + kb + 8 + 2 * i);
      }
    }
    #pragma unroll
    for (int nj = 0; nj < 4; ++nj) {
      FragB b;
      const u16* bp = &b_s[cur][(nj * 16 + lm) * STRA];
      int kb2 = lh * 16;
      #pragma unroll
      for (int i = 0; i < 8; ++i) b.u[i] = *(const u32*)(bp + kb2 + 2 * i);
      #pragma unroll
      for (int mt = 0; mt < 2; ++mt)
        acc[mt][nj] = __builtin_amdgcn_wmma_f32_16x16x32_bf16(
            false, a[mt].v, false, b.v, (short)0, acc[mt][nj], false, false);
    }
    wait_async0();
    __syncthreads();
  }

  #pragma unroll
  for (int mt = 0; mt < 2; ++mt) {
    #pragma unroll
    for (int nj = 0; nj < 4; ++nj) {
      #pragma unroll
      for (int i = 0; i < 8; ++i) {
        int gm = m0 + wv * 32 + mt * 16 + i + ((lane & 16) ? 8 : 0);
        int gc = n0 + nj * 16 + lm;
        out[(size_t)gm * DIM + gc] = acc[mt][nj][i] + bias[gc];
      }
    }
  }
}

// ---------------------------------------------------------------------------
extern "C" void kernel_launch(void* const* d_in, const int* in_sizes, int n_in,
                              void* d_out, int out_size, void* d_ws, size_t ws_size,
                              hipStream_t stream) {
  const float* x      = (const float*)d_in[0];
  const float* mask   = (const float*)d_in[1];
  const float* qkv_w  = (const float*)d_in[2];
  const float* qkv_b  = (const float*)d_in[3];
  const float* bt     = (const float*)d_in[4];
  const float* proj_w = (const float*)d_in[5];
  const float* proj_b = (const float*)d_in[6];
  const int*   ridx   = (const int*)d_in[7];
  float* out = (float*)d_out;

  char* ws = (char*)d_ws;
  const size_t QKV_BYTES  = (size_t)BWIN * HEADS * 3 * NTOK * HD * sizeof(u16);
  const size_t AO_BYTES   = (size_t)MTOT * DIM * sizeof(u16);
  const size_t BIAS_BYTES = (size_t)HEADS * NTOK * NTOK * sizeof(float);
  const size_t XB_BYTES   = (size_t)MTOT * DIM * sizeof(u16);
  const size_t WBT_BYTES  = (size_t)DIM * (3 * DIM) * sizeof(u16);

  u16*   qkv_ws = (u16*)ws;
  u16*   ao     = (u16*)(ws + QKV_BYTES);
  float* bias_h = (float*)(ws + QKV_BYTES + AO_BYTES);
  u16*   xb     = (u16*)(ws + QKV_BYTES + AO_BYTES + BIAS_BYTES);
  u16*   wbT    = (u16*)(ws + QKV_BYTES + AO_BYTES + BIAS_BYTES + XB_BYTES);
  u16*   pwbT   = (u16*)(ws + QKV_BYTES + AO_BYTES + BIAS_BYTES + XB_BYTES + WBT_BYTES);

  // pre-convert inputs to bf16 (x elementwise; weights transposed)
  int xn4 = (MTOT * DIM) / 4;
  cvt_bf16_kernel<<<(xn4 + 255) / 256, 256, 0, stream>>>(x, xb, xn4);
  cvtT_bf16_kernel<<<dim3(DIM / 32, (3 * DIM) / 32), 256, 0, stream>>>(qkv_w, wbT, DIM, 3 * DIM);
  cvtT_bf16_kernel<<<dim3(DIM / 32, DIM / 32), 256, 0, stream>>>(proj_w, pwbT, DIM, DIM);
  bias_mat_kernel<<<(HEADS * NTOK * NTOK + 255) / 256, 256, 0, stream>>>(bt, ridx, bias_h);

  qkv_gemm_kernel<<<(MTOT / 128) * (3 * DIM / 64), 128, 0, stream>>>(xb, wbT, qkv_b, qkv_ws);
  attn_kernel<<<BWIN * HEADS, 128, 0, stream>>>(qkv_ws, bias_h, mask, ao);
  proj_gemm_kernel<<<(MTOT / 128) * (DIM / 64), 128, 0, stream>>>(ao, pwbT, proj_b, out);
}
